// Attention_24807731102367
// MI455X (gfx1250) — matrix operand order repaired and checked by reading, not compiled, not run on hardware
//
#include <hip/hip_runtime.h>

// Problem constants (match reference)
#define DIMK   1024
#define NHEAD  16
#define DHEAD  64
#define SEQ    2048
#define BATCH  4
#define MROWS  (BATCH * SEQ)   // 8192 flattened (b, n) rows

typedef __attribute__((ext_vector_type(16))) _Float16 v16h;
typedef __attribute__((ext_vector_type(8)))  _Float16 v8h;
typedef __attribute__((ext_vector_type(8)))  float    v8f;
typedef __attribute__((ext_vector_type(4)))  unsigned su4;
typedef __attribute__((ext_vector_type(8)))  unsigned su8;

__device__ __forceinline__ v8f wmma16x16x32(v16h a, v16h b, v8f c) {
  return __builtin_amdgcn_wmma_f32_16x16x32_f16(
      /*neg_a=*/false, a, /*neg_b=*/false, b,
      /*c_mod=*/(short)0, c, /*reuse_a=*/false, /*reuse_b=*/false);
}

// ---- CDNA5 async global->LDS copy (ASYNCcnt-tracked), 16 bytes per lane ---
__device__ __forceinline__ void async_cp16(const void* gaddr, void* lds_ptr) {
  unsigned loff = (unsigned)(unsigned long long)lds_ptr;   // LDS byte offset
  unsigned long long ga = (unsigned long long)gaddr;
  asm volatile("global_load_async_to_lds_b128 %0, %1, off"
               :: "v"(loff), "v"(ga) : "memory");
}
__device__ __forceinline__ void async_wait0() {
  asm volatile("s_wait_asynccnt 0x0" ::: "memory");
}

// ---- CDNA5 Tensor Data Mover: 2-D tile DMA global->LDS (TENSORcnt) -------
// D# group0: count=1 | lds_addr | global_addr[56:0] | type=2 ("image")
// D# group1: workgroup_mask=0, data_size=1 (2B), pad cfg, dims/strides
__device__ __forceinline__ void tdm_load_2d(const void* gaddr, void* ldsaddr,
                                            unsigned g1w0,
                                            unsigned tdim0, unsigned tdim1,
                                            unsigned tile0, unsigned tile1,
                                            unsigned stride0) {
  unsigned long long ga = (unsigned long long)gaddr;
  su4 g0;
  g0[0] = 1u;                                              // count=1 (valid)
  g0[1] = (unsigned)(unsigned long long)ldsaddr;           // lds_addr (bytes)
  g0[2] = (unsigned)ga;                                    // global_addr[31:0]
  g0[3] = (unsigned)((ga >> 32) & 0x01FFFFFFu) | 0x80000000u; // ga[56:32]|type=2
  su8 g1;
  g1[0] = g1w0;                                            // dsize/pad config
  g1[1] = (tdim0 & 0xFFFFu) << 16;                         // tensor_dim0[15:0]
  g1[2] = ((tdim0 >> 16) & 0xFFFFu) | ((tdim1 & 0xFFFFu) << 16);
  g1[3] = ((tdim1 >> 16) & 0xFFFFu) | (tile0 << 16);       // tile_dim0
  g1[4] = tile1;                                           // tile_dim1 (dim2=0)
  g1[5] = stride0;                                         // dim0_stride[31:0]
  g1[6] = 0;
  g1[7] = 0;
  asm volatile("tensor_load_to_lds %0, %1" :: "s"(g0), "s"(g1) : "memory");
}

// data_size=2B | pad_enable | pad_interval code | pad_amount code(4 DWORDs)
#define G1_K ((1u << 16) | (1u << 20) | (4u << 22) | (3u << 25)) // 32DW rows -> pitch 72h
#define G1_V ((1u << 16) | (1u << 20) | (3u << 22) | (3u << 25)) // 16DW rows -> pitch 40h

// ---- fragment loaders (contiguous-k layouts, wave32) ------------------------
// A 16x32 f16, row-major src, stride ld (elements).
__device__ __forceinline__ v16h load_a_f16(const _Float16* p, int ld) {
  const int lane = threadIdx.x & 31;
  const int row  = lane & 15;
  const int kb   = (lane >> 4) << 4;           // 0 or 16
  const _Float16* r = p + (size_t)row * ld;
  v16h a;
#pragma unroll
  for (int j = 0; j < 8; ++j) a[j]     = r[kb + j];
#pragma unroll
  for (int j = 0; j < 8; ++j) a[8 + j] = r[kb + 8 + j];
  return a;
}

// B 32x16 with k CONTIGUOUS per lane: B[k][col] = p[col*ld + k]
__device__ __forceinline__ v16h load_bk_f16(const _Float16* p, int ld) {
  const int lane = threadIdx.x & 31;
  const int col  = lane & 15;
  const int kb   = (lane >> 4) << 4;           // 0 or 16
  const _Float16* r = p + (size_t)col * ld + kb;
  v16h b;
#pragma unroll
  for (int j = 0; j < 16; ++j) b[j] = r[j];
  return b;
}

// ==== stage 1: projection  dst(f16) = A(f32)[m,k] @ W(f32)[k,n] ============
// VT=0: dst layout [b,h,n,d];  VT=1: dst layout [b,h,d,n] (V transposed)
// grid (N/64, M/64), block 128 (4 waves); block tile 64x64, K-step 64.
#define SPITCH 72   // 64 halfs + 8 pad

template <int VT>
__global__ __launch_bounds__(128) void proj_kernel(
    const float* __restrict__ A, const float* __restrict__ W,
    _Float16* __restrict__ dst) {
  __shared__ _Float16 As[64 * SPITCH];   // 64 rows x 64 k
  __shared__ _Float16 Ws[64 * SPITCH];   // W^T: 64 cols x 64 k
  const int tid  = threadIdx.x;
  const int wave = tid >> 5;
  const int lane = tid & 31;
  const int row0 = blockIdx.y * 64;
  const int col0 = blockIdx.x * 64;

  v8f acc[4] = {};
  for (int k0 = 0; k0 < DIMK; k0 += 64) {
    {   // stage A tile: f32 -> f16, coalesced 32-float row chunks
      const int r = tid >> 1, kq = (tid & 1) * 32;
      const float* g = A + (size_t)(row0 + r) * DIMK + k0 + kq;
      _Float16* s = As + r * SPITCH + kq;
#pragma unroll
      for (int j = 0; j < 32; ++j) s[j] = (_Float16)g[j];
    }
    {   // stage W tile transposed: coalesced global rows, strided LDS writes
      const int kr = tid >> 1, cb = (tid & 1) * 32;
      const float* g = W + (size_t)(k0 + kr) * DIMK + col0 + cb;
#pragma unroll
      for (int j = 0; j < 32; ++j) Ws[(cb + j) * SPITCH + kr] = (_Float16)g[j];
    }
    __syncthreads();
#pragma unroll
    for (int kc = 0; kc < 64; kc += 32) {
      v16h a = load_a_f16(As + wave * 16 * SPITCH + kc, SPITCH);
#pragma unroll
      for (int t = 0; t < 4; ++t) {
        v16h b = load_bk_f16(Ws + t * 16 * SPITCH + kc, SPITCH);
        acc[t] = wmma16x16x32(a, b, acc[t]);
      }
    }
    __syncthreads();
  }

  const int c  = lane & 15;
  const int rb = (lane >> 4) << 3;
  const int wrow0 = row0 + wave * 16;
#pragma unroll
  for (int t = 0; t < 4; ++t) {
    const int ncol = col0 + t * 16 + c;
    const int head = ncol >> 6, dl = ncol & 63;
#pragma unroll
    for (int r = 0; r < 8; ++r) {
      const int m  = wrow0 + rb + r;
      const int bb = m >> 11, nn = m & 2047;
      const size_t idx = VT
          ? (((size_t)bb * NHEAD + head) * DHEAD + dl) * SEQ + nn
          : (((size_t)bb * NHEAD + head) * SEQ + nn) * DHEAD + dl;
      dst[idx] = (_Float16)acc[t][r];
    }
  }
}

// ==== stage 2: flash attention (transposed S/O), TDM-staged K/V^T ==========
// grid (SEQ/64, BATCH*NHEAD), block 128 (4 waves), 16 query COLS per wave.
// S^T = K @ Q^T: each lane owns one query column -> per-lane scalar softmax
// state; one xor-16 shuffle per reduction per 32-key tile.
#define KPITCH 72   // 64 halfs + 8 pad (from TDM pad cfg)
#define VPITCH 40   // 32 halfs + 8 pad (from TDM pad cfg)
#define PPITCH 40   // P^T tile: 32 halfs + 8 pad

// Process one 32-key tile from LDS buffers ks/vs.
__device__ __forceinline__ void attn_tile(
    const _Float16* __restrict__ ks, const _Float16* __restrict__ vs,
    v16h bq0, v16h bq1, v8f* o, float& mi, float& li,
    _Float16* pstore /* &plds[wave][c][rb] */,
    const _Float16* pbase /* &plds[wave][0][0] */) {
  // S^T(32x16) = K_tile(32x64) @ Q^T(64x16): batch all fragment loads first
  const v16h ak0 = load_a_f16(ks + 0 * 16 * KPITCH + 0,  KPITCH);
  const v16h ak1 = load_a_f16(ks + 0 * 16 * KPITCH + 32, KPITCH);
  const v16h ak2 = load_a_f16(ks + 1 * 16 * KPITCH + 0,  KPITCH);
  const v16h ak3 = load_a_f16(ks + 1 * 16 * KPITCH + 32, KPITCH);
  v8f s0 = {}, s1 = {};
  s0 = wmma16x16x32(ak0, bq0, s0);
  s0 = wmma16x16x32(ak1, bq1, s0);
  s1 = wmma16x16x32(ak2, bq0, s1);
  s1 = wmma16x16x32(ak3, bq1, s1);

  // V^T A-fragments are independent of softmax: issue loads now, overlap
  const v16h av0 = load_a_f16(vs + 0 * 16 * VPITCH, VPITCH);
  const v16h av1 = load_a_f16(vs + 1 * 16 * VPITCH, VPITCH);
  const v16h av2 = load_a_f16(vs + 2 * 16 * VPITCH, VPITCH);
  const v16h av3 = load_a_f16(vs + 3 * 16 * VPITCH, VPITCH);

  // online softmax over keys for this lane's query column
  const float scale = 0.125f;                 // 1/sqrt(64)
  float cand = -1e30f;
#pragma unroll
  for (int r = 0; r < 8; ++r) {
    s0[r] *= scale; s1[r] *= scale;
    cand = fmaxf(cand, fmaxf(s0[r], s1[r]));
  }
  cand = fmaxf(cand, __shfl_xor(cand, 16, 32));     // other 16 keys
  const float nmax = fmaxf(mi, cand);
  const float al   = __expf(mi - nmax);
  float ps = 0.f;
  v8h pk0, pk1;
#pragma unroll
  for (int r = 0; r < 8; ++r) {
    const float p0 = __expf(s0[r] - nmax);
    const float p1 = __expf(s1[r] - nmax);
    ps += p0 + p1;
    pk0[r] = (_Float16)p0;
    pk1[r] = (_Float16)p1;
  }
  ps += __shfl_xor(ps, 16, 32);
  li = li * al + ps;
  mi = nmax;
#pragma unroll
  for (int t = 0; t < 4; ++t)
#pragma unroll
    for (int r = 0; r < 8; ++r) o[t][r] *= al;

  // P^T tile: 8 contiguous halfs per chunk -> packed ds_store_b128
  *(v8h*)(pstore)      = pk0;
  *(v8h*)(pstore + 16) = pk1;

  // O^T(64x16) += V^T(64x32) @ P^T(32x16)
  const v16h bp = load_bk_f16(pbase, PPITCH);
  o[0] = wmma16x16x32(av0, bp, o[0]);
  o[1] = wmma16x16x32(av1, bp, o[1]);
  o[2] = wmma16x16x32(av2, bp, o[2]);
  o[3] = wmma16x16x32(av3, bp, o[3]);
}

__global__ __launch_bounds__(128) void attn_kernel(
    const _Float16* __restrict__ Qf, const _Float16* __restrict__ Kf,
    const _Float16* __restrict__ Vt, _Float16* __restrict__ Of) {
  __shared__ _Float16 Ks[2][32 * KPITCH];
  __shared__ _Float16 Vs[2][64 * VPITCH];
  __shared__ _Float16 plds[4][16][PPITCH];   // per-wave P^T tile [q][m]

  const int tid  = threadIdx.x;
  const int wave = tid >> 5;
  const int lane = tid & 31;
  const int bh   = blockIdx.y;
  const int q0   = blockIdx.x * 64 + wave * 16;

  const _Float16* Q = Qf + (size_t)bh * SEQ * DHEAD;
  const _Float16* K = Kf + (size_t)bh * SEQ * DHEAD;
  const _Float16* V = Vt + (size_t)bh * DHEAD * SEQ;   // [d][n]

  // Q^T as B-operand fragments (resident for whole loop)
  const v16h bq0 = load_bk_f16(Q + (size_t)q0 * DHEAD + 0,  DHEAD);
  const v16h bq1 = load_bk_f16(Q + (size_t)q0 * DHEAD + 32, DHEAD);

  v8f o[4] = {};                      // O^T: rows d = t*16+rb+r, col = query
  float mi = -1e30f, li = 0.f;        // per-lane (per-query) softmax state

  const int c  = lane & 15;
  const int rb = (lane >> 4) << 3;
  _Float16* pstore = &plds[wave][c][rb];
  const _Float16* pbase = &plds[wave][0][0];

  if (wave == 0) {   // one TDM descriptor pair per tile, single issuing wave
    tdm_load_2d(K, Ks[0], G1_K, DHEAD, SEQ, 64, 32, DHEAD);
    tdm_load_2d(V, Vs[0], G1_V, SEQ, DHEAD, 32, 64, SEQ);
  }
  // Unrolled x2: explicit double-buffer phases, no pointer selects
  for (int mt = 0; mt < SEQ; mt += 64) {
    // -------- phase A: compute buffer 0, DMA (mt+32) into buffer 1 --------
    if (wave == 0) __builtin_amdgcn_s_wait_tensorcnt(0);
    __syncthreads();
    if (wave == 0) {   // mt+32 <= SEQ-32 always inside this loop
      tdm_load_2d(K + (size_t)(mt + 32) * DHEAD, Ks[1],
                  G1_K, DHEAD, SEQ, 64, 32, DHEAD);
      tdm_load_2d(V + (mt + 32), Vs[1], G1_V, SEQ, DHEAD, 32, 64, SEQ);
    }
    attn_tile(Ks[0], Vs[0], bq0, bq1, o, mi, li, pstore, pbase);

    // -------- phase B: compute buffer 1, DMA (mt+64) into buffer 0 --------
    if (wave == 0) __builtin_amdgcn_s_wait_tensorcnt(0);
    __syncthreads();
    if (wave == 0 && mt + 64 < SEQ) {
      tdm_load_2d(K + (size_t)(mt + 64) * DHEAD, Ks[0],
                  G1_K, DHEAD, SEQ, 64, 32, DHEAD);
      tdm_load_2d(V + (mt + 64), Vs[0], G1_V, SEQ, DHEAD, 32, 64, SEQ);
    }
    attn_tile(Ks[1], Vs[1], bq0, bq1, o, mi, li, pstore, pbase);
  }

  // epilogue: lane owns query row (q0+c); packed contiguous b128 stores
  const float inv = 1.f / li;
  const int bb = bh >> 4, hh = bh & 15;
  _Float16* orow = Of + ((size_t)bb * SEQ + q0 + c) * (NHEAD * DHEAD) + hh * DHEAD;
#pragma unroll
  for (int t = 0; t < 4; ++t) {
    v8h ov;
#pragma unroll
    for (int r = 0; r < 8; ++r) ov[r] = (_Float16)(o[t][r] * inv);
    *(v8h*)&orow[t * 16 + rb] = ov;
  }
}

// ==== stage 3: out(f32) = attn_out(f16)[m,k] @ W_o(f32)[k,n] + b_o =========
__global__ __launch_bounds__(128) void outproj_kernel(
    const _Float16* __restrict__ A, const float* __restrict__ W,
    const float* __restrict__ bias, float* __restrict__ out) {
  __shared__ _Float16 As[64 * SPITCH];
  __shared__ _Float16 Ws[64 * SPITCH];
  const int tid  = threadIdx.x;
  const int wave = tid >> 5;
  const int lane = tid & 31;
  const int row0 = blockIdx.y * 64;
  const int col0 = blockIdx.x * 64;

  v8f acc[4] = {};
  for (int k0 = 0; k0 < DIMK; k0 += 64) {
    // async-stage A tile (already f16): 512 x 16B chunks, 4 per thread
#pragma unroll
    for (int i = 0; i < 4; ++i) {
      const int ch = tid * 4 + i;
      const int r = ch >> 3, ci = ch & 7;
      async_cp16(A + (size_t)(row0 + r) * DIMK + k0 + ci * 8,
                 As + r * SPITCH + ci * 8);
    }
    {   // stage W transposed with f32->f16 conversion
      const int kr = tid >> 1, cb = (tid & 1) * 32;
      const float* g = W + (size_t)(k0 + kr) * DIMK + col0 + cb;
#pragma unroll
      for (int j = 0; j < 32; ++j) Ws[(cb + j) * SPITCH + kr] = (_Float16)g[j];
    }
    async_wait0();
    __syncthreads();
#pragma unroll
    for (int kc = 0; kc < 64; kc += 32) {
      v16h a = load_a_f16(As + wave * 16 * SPITCH + kc, SPITCH);
#pragma unroll
      for (int t = 0; t < 4; ++t) {
        v16h b = load_bk_f16(Ws + t * 16 * SPITCH + kc, SPITCH);
        acc[t] = wmma16x16x32(a, b, acc[t]);
      }
    }
    __syncthreads();
  }

  const int c  = lane & 15;
  const int rb = (lane >> 4) << 3;
  const int wrow0 = row0 + wave * 16;
#pragma unroll
  for (int t = 0; t < 4; ++t) {
    const int ncol = col0 + t * 16 + c;
    const float bo = bias[ncol];
#pragma unroll
    for (int r = 0; r < 8; ++r) {
      const int m = wrow0 + rb + r;
      out[(size_t)m * DIMK + ncol] = acc[t][r] + bo;
    }
  }
}

extern "C" void kernel_launch(void* const* d_in, const int* in_sizes, int n_in,
                              void* d_out, int out_size, void* d_ws, size_t ws_size,
                              hipStream_t stream) {
  const float* x   = (const float*)d_in[0];
  const float* ctx = (const float*)d_in[1];
  const float* Wq  = (const float*)d_in[2];
  const float* Wk  = (const float*)d_in[3];
  const float* Wv  = (const float*)d_in[4];
  const float* Wo  = (const float*)d_in[5];
  const float* bo  = (const float*)d_in[6];
  float* out = (float*)d_out;

  const size_t tsz = (size_t)MROWS * DIMK;     // 8192*1024 f16 elements
  _Float16* qf = (_Float16*)d_ws;              // [b,h,n,d]
  _Float16* kf = qf + tsz;                     // [b,h,n,d]
  _Float16* vf = kf + tsz;                     // [b,h,d,n]  (transposed)
  _Float16* af = vf + tsz;                     // [b,n,h*d]

  dim3 blk(128);
  dim3 gproj(DIMK / 64, MROWS / 64);           // (16, 128)
  proj_kernel<0><<<gproj, blk, 0, stream>>>(x,   Wq, qf);
  proj_kernel<0><<<gproj, blk, 0, stream>>>(ctx, Wk, kf);
  proj_kernel<1><<<gproj, blk, 0, stream>>>(ctx, Wv, vf);

  dim3 gattn(SEQ / 64, BATCH * NHEAD);         // (32, 64)
  attn_kernel<<<gattn, blk, 0, stream>>>(qf, kf, vf, af);

  outproj_kernel<<<gproj, blk, 0, stream>>>(af, Wo, bo, out);
}
